// AnomalyDetector_38293928411575
// MI455X (gfx1250) — compile-verified, matrix operand order checked by reading, not run on hardware
//
#include <hip/hip_runtime.h>

// ---------- types ----------
typedef _Float16 v16h __attribute__((ext_vector_type(16)));
typedef float    v8f  __attribute__((ext_vector_type(8)));
typedef unsigned int u32x4 __attribute__((ext_vector_type(4)));

union Frag16 { v16h h; u32x4 q[2]; };

#define LSTM_T   32
#define LSTM_F   64
#define LSTM_H   256
#define LSTM_K   320   // F + H
#define BLK_ROWS 64
#define A_STRIDE 328   // f16 elems per LDS row (padded, 16B-aligned rows)

__device__ __forceinline__ float sigf(float x) { return 1.0f / (1.0f + __expf(-x)); }
// branchless tanh on the v_exp_f32 path; saturates to +/-1 at extremes (no NaN)
__device__ __forceinline__ float tanh_fast(float x) {
  return 2.0f / (1.0f + __expf(-2.0f * x)) - 1.0f;
}

// ---------- prep: pack [w_ih | w_hh] row-major f16 (B columns k-contiguous), bias sum ----------
__global__ void pack_lstm(const float* __restrict__ w_ih, const float* __restrict__ w_hh,
                          const float* __restrict__ b_ih, const float* __restrict__ b_hh,
                          _Float16* __restrict__ wcat, float* __restrict__ bias) {
  int i = blockIdx.x * 256 + threadIdx.x;
  if (i < 1024 * LSTM_K) {
    int g = i / LSTM_K, k = i % LSTM_K;
    float v = (k < LSTM_F) ? w_ih[g * LSTM_F + k] : w_hh[g * LSTM_H + (k - LSTM_F)];
    wcat[i] = (_Float16)v;
  }
  if (i < 1024) bias[i] = b_ih[i] + b_hh[i];
}

// wT[p*K + k] = w[k*P + p]  (transpose + f16)
__global__ void pack_transpose(const float* __restrict__ w, _Float16* __restrict__ wT,
                               int K, int P) {
  int i = blockIdx.x * 256 + threadIdx.x;
  if (i < K * P) { int p = i / K, k = i % K; wT[i] = (_Float16)w[k * P + p]; }
}

// ---------- LSTM: persistent per-node, h/c in regs+LDS, weights streamed from L2 ----------
__global__ __launch_bounds__(256)
void lstm_kernel(const float* __restrict__ x, const _Float16* __restrict__ wcat,
                 const float* __restrict__ bias, _Float16* __restrict__ hout, int N) {
  extern __shared__ _Float16 Abuf[];               // 2 panels of [BLK_ROWS x A_STRIDE]
  const int tid  = threadIdx.x;
  const int lane = tid & 31;
  const int wid  = tid >> 5;
  const int mt   = wid & 3;                        // row tile (16 rows)
  const int ch   = wid >> 2;                       // column half of each 64-hidden chunk
  const int row0 = blockIdx.x * BLK_ROWS;
  const int lm   = lane & 15;
  const int lh   = lane >> 4;

  // biases (loop-invariant; bv[hc][gate][ntile], added at elementwise stage)
  float bv[4][4][2];
#pragma unroll
  for (int hc = 0; hc < 4; ++hc)
#pragma unroll
    for (int g = 0; g < 4; ++g)
#pragma unroll
      for (int nt = 0; nt < 2; ++nt)
        bv[hc][g][nt] = bias[256 * g + 64 * hc + 32 * ch + 16 * nt + lm];

  // zero h-region of panel 0
  for (int i = tid; i < BLK_ROWS * LSTM_H; i += 256) {
    int r = i >> 8, k = i & 255;
    Abuf[r * A_STRIDE + LSTM_F + k] = (_Float16)0.0f;
  }
  // load x_0 into panel 0 (each thread: 16 consecutive f32 -> f16)
  {
    int r = tid >> 2, f0 = (tid & 3) * 16;
    int node = row0 + r; if (node >= N) node = N - 1;
    const float4* xp = (const float4*)(x + ((size_t)node * LSTM_T + 0) * LSTM_F + f0);
#pragma unroll
    for (int q = 0; q < 4; ++q) {
      float4 xv = xp[q];
      _Float16* d = &Abuf[r * A_STRIDE + f0 + q * 4];
      d[0] = (_Float16)xv.x; d[1] = (_Float16)xv.y; d[2] = (_Float16)xv.z; d[3] = (_Float16)xv.w;
    }
  }
  __syncthreads();

  float cr[4][2][8];
#pragma unroll
  for (int hc = 0; hc < 4; ++hc)
#pragma unroll
    for (int nt = 0; nt < 2; ++nt)
#pragma unroll
      for (int v = 0; v < 8; ++v) cr[hc][nt][v] = 0.0f;

  int cur = 0;
#pragma unroll 1
  for (int t = 0; t < LSTM_T; ++t) {
    const int nxt = cur ^ 1;
    _Float16* Acur = Abuf + cur * (BLK_ROWS * A_STRIDE);
    _Float16* Anxt = Abuf + nxt * (BLK_ROWS * A_STRIDE);

#pragma unroll
    for (int hc = 0; hc < 4; ++hc) {
#pragma unroll
      for (int nt = 0; nt < 2; ++nt) {
        // 16 gate columns per gate group; 4 accumulators (i,f,g,o)
        v8f acc0 = {}, acc1 = {}, acc2 = {}, acc3 = {};
        const int jcol = 64 * hc + 32 * ch + 16 * nt + lm;  // hidden-unit column
        const _Float16* b0p = wcat + (size_t)(0 * 256 + jcol) * LSTM_K + lh * 8;
        const _Float16* b1p = wcat + (size_t)(1 * 256 + jcol) * LSTM_K + lh * 8;
        const _Float16* b2p = wcat + (size_t)(2 * 256 + jcol) * LSTM_K + lh * 8;
        const _Float16* b3p = wcat + (size_t)(3 * 256 + jcol) * LSTM_K + lh * 8;
        const _Float16* arow = Acur + (mt * 16 + lm) * A_STRIDE + lh * 8;

#pragma unroll 1      // keep rolled: B addresses depend on kt -> no LICM hoist/spill
        for (int kt = 0; kt < LSTM_K / 32; ++kt) {
          const int kb = kt * 32;
          Frag16 a, b0, b1, b2, b3;
          a.q[0]  = *(const u32x4*)(arow + kb);
          a.q[1]  = *(const u32x4*)(arow + kb + 16);
          b0.q[0] = *(const u32x4*)(b0p + kb);
          b0.q[1] = *(const u32x4*)(b0p + kb + 16);
          b1.q[0] = *(const u32x4*)(b1p + kb);
          b1.q[1] = *(const u32x4*)(b1p + kb + 16);
          b2.q[0] = *(const u32x4*)(b2p + kb);
          b2.q[1] = *(const u32x4*)(b2p + kb + 16);
          b3.q[0] = *(const u32x4*)(b3p + kb);
          b3.q[1] = *(const u32x4*)(b3p + kb + 16);
          acc0 = __builtin_amdgcn_wmma_f32_16x16x32_f16(false, a.h, false, b0.h, (short)0, acc0, false, false);
          acc1 = __builtin_amdgcn_wmma_f32_16x16x32_f16(false, a.h, false, b1.h, (short)0, acc1, false, false);
          acc2 = __builtin_amdgcn_wmma_f32_16x16x32_f16(false, a.h, false, b2.h, (short)0, acc2, false, false);
          acc3 = __builtin_amdgcn_wmma_f32_16x16x32_f16(false, a.h, false, b3.h, (short)0, acc3, false, false);
        }

        // elementwise LSTM update; write h(f16) into next panel
#pragma unroll
        for (int v = 0; v < 8; ++v) {
          float iv = sigf(acc0[v] + bv[hc][0][nt]);
          float fv = sigf(acc1[v] + bv[hc][1][nt]);
          float gv = tanh_fast(acc2[v] + bv[hc][2][nt]);
          float ov = sigf(acc3[v] + bv[hc][3][nt]);
          float c  = fv * cr[hc][nt][v] + iv * gv;
          cr[hc][nt][v] = c;
          float h  = ov * tanh_fast(c);
          int r = mt * 16 + v + 8 * lh;
          int j = 64 * hc + 32 * ch + 16 * nt + lm;
          _Float16 hh = (_Float16)h;
          Anxt[r * A_STRIDE + LSTM_F + j] = hh;
          if (t == LSTM_T - 1) {
            int node = row0 + r;
            if (node < N) hout[(size_t)node * LSTM_H + j] = hh;
          }
        }
      }
    }

    // stage x_{t+1} into next panel
    if (t + 1 < LSTM_T) {
      int r = tid >> 2, f0 = (tid & 3) * 16;
      int node = row0 + r; if (node >= N) node = N - 1;
      const float4* xp = (const float4*)(x + ((size_t)node * LSTM_T + (t + 1)) * LSTM_F + f0);
#pragma unroll
      for (int q = 0; q < 4; ++q) {
        float4 xv = xp[q];
        _Float16* d = &Anxt[r * A_STRIDE + f0 + q * 4];
        d[0] = (_Float16)xv.x; d[1] = (_Float16)xv.y; d[2] = (_Float16)xv.z; d[3] = (_Float16)xv.w;
      }
    }
    __syncthreads();
    cur = nxt;
  }
}

// ---------- generic WMMA GEMM: C[n x P] = A_f16[n x K] * BT_f16[P x K]^T ----------
__global__ __launch_bounds__(128)
void gemm_f16(const _Float16* __restrict__ A, const _Float16* __restrict__ BT,
              float* __restrict__ C, int nrows, int K, int P) {
  const int lane = threadIdx.x & 31;
  const int wid  = threadIdx.x >> 5;
  const int mtile = blockIdx.x * 4 + wid;
  const int pt    = blockIdx.y;
  const int lm = lane & 15, lh = lane >> 4;
  int row  = mtile * 16 + lm;
  int rowc = row < nrows ? row : nrows - 1;
  int col  = pt * 16 + lm;
  v8f acc = {};
  for (int kt = 0; kt < K / 32; ++kt) {
    int kb = kt * 32 + lh * 8;
    Frag16 a, b;
    const _Float16* ap = A + (size_t)rowc * K + kb;
    const _Float16* bp = BT + (size_t)col * K + kb;
    a.q[0] = *(const u32x4*)(ap);  a.q[1] = *(const u32x4*)(ap + 16);
    b.q[0] = *(const u32x4*)(bp);  b.q[1] = *(const u32x4*)(bp + 16);
    acc = __builtin_amdgcn_wmma_f32_16x16x32_f16(false, a.h, false, b.h, (short)0, acc, false, false);
  }
#pragma unroll
  for (int v = 0; v < 8; ++v) {
    int r = mtile * 16 + v + 8 * lh;
    if (r < nrows) C[(size_t)r * P + pt * 16 + lm] = acc[v];
  }
}

// ---------- GCN aggregation ----------
__global__ void deg_init(float* __restrict__ deg, int n) {
  int i = blockIdx.x * 256 + threadIdx.x; if (i < n) deg[i] = 1.0f;   // self loop
}
__global__ void deg_count(const int* __restrict__ dst, float* __restrict__ deg, int E) {
  int e = blockIdx.x * 256 + threadIdx.x; if (e < E) atomicAdd(&deg[dst[e]], 1.0f);
}
__global__ void dinv_k(const float* __restrict__ deg, float* __restrict__ dinv, int n) {
  int i = blockIdx.x * 256 + threadIdx.x; if (i < n) dinv[i] = rsqrtf(deg[i]);
}
// acc[v,:] = dinv[v] * g[v,:]  (self-loop term; final pass multiplies dinv[v] again)
__global__ void agg_init(const float* __restrict__ g, const float* __restrict__ dinv,
                         float* __restrict__ acc, long long total, int P) {
  long long i = (long long)blockIdx.x * 256 + threadIdx.x;
  if (i < total) acc[i] = dinv[(int)(i / P)] * g[i];
}
__global__ void agg_edge(const int* __restrict__ src, const int* __restrict__ dst,
                         const float* __restrict__ dinv, const float* __restrict__ g,
                         float* __restrict__ acc, int E, int P) {
  long long gid = (long long)blockIdx.x * 256 + threadIdx.x;
  int chunks = P >> 3;
  long long e = gid / chunks;
  if (e >= E) return;
  int jc = (int)(gid - e * chunks) * 8;
  int s = src[e], d = dst[e];
  float w = dinv[s];
  const float* gs = g + (size_t)s * P + jc;
  float* ad = acc + (size_t)d * P + jc;
#pragma unroll
  for (int q = 0; q < 8; ++q) atomicAdd(&ad[q], w * gs[q]);
}
__global__ void finalize_f16(const float* __restrict__ acc, const float* __restrict__ dinv,
                             const float* __restrict__ b, _Float16* __restrict__ out,
                             long long total, int P) {
  long long i = (long long)blockIdx.x * 256 + threadIdx.x;
  if (i >= total) return;
  float r = dinv[(int)(i / P)] * acc[i] + b[(int)(i % P)];
  out[i] = (_Float16)(r > 0.0f ? r : 0.0f);
}
__global__ void finalize_f32(const float* __restrict__ acc, const float* __restrict__ dinv,
                             const float* __restrict__ b, float* __restrict__ out,
                             long long total, int P) {
  long long i = (long long)blockIdx.x * 256 + threadIdx.x;
  if (i >= total) return;
  float r = dinv[(int)(i / P)] * acc[i] + b[(int)(i % P)];
  out[i] = r > 0.0f ? r : 0.0f;
}
__global__ void classifier_k(const float* __restrict__ feat, const float* __restrict__ wc,
                             const float* __restrict__ bc, float* __restrict__ out, int n) {
  int v = blockIdx.x * 256 + threadIdx.x;
  if (v >= n) return;
  float z0 = bc[0], z1 = bc[1];
  const float* f = feat + (size_t)v * 128;
#pragma unroll 4
  for (int j = 0; j < 128; ++j) { float fv = f[j]; z0 += fv * wc[2 * j]; z1 += fv * wc[2 * j + 1]; }
  float m = fmaxf(z0, z1);
  float e0 = __expf(z0 - m), e1 = __expf(z1 - m);
  float inv = 1.0f / (e0 + e1);
  out[2 * v] = e0 * inv; out[2 * v + 1] = e1 * inv;
}

// ---------- launch ----------
extern "C" void kernel_launch(void* const* d_in, const int* in_sizes, int n_in,
                              void* d_out, int out_size, void* d_ws, size_t ws_size,
                              hipStream_t stream) {
  const float* x    = (const float*)d_in[0];
  const int*   ei   = (const int*)d_in[1];
  const float* w_ih = (const float*)d_in[2];
  const float* w_hh = (const float*)d_in[3];
  const float* b_ih = (const float*)d_in[4];
  const float* b_hh = (const float*)d_in[5];
  const float* w1   = (const float*)d_in[6];
  const float* b1   = (const float*)d_in[7];
  const float* w2   = (const float*)d_in[8];
  const float* b2   = (const float*)d_in[9];
  const float* wc   = (const float*)d_in[10];
  const float* bc   = (const float*)d_in[11];

  const int N = in_sizes[0] / (LSTM_T * LSTM_F);
  const int E = in_sizes[1] / 2;
  const int* src = ei;
  const int* dst = ei + E;

  char* ws = (char*)d_ws;
  size_t off = 0;
  auto alloc = [&](size_t bytes) { size_t o = off; off += (bytes + 255) & ~(size_t)255; return o; };
  _Float16* wcat = (_Float16*)(ws + alloc(1024 * LSTM_K * 2));
  float*    bias = (float*)   (ws + alloc(1024 * 4));
  _Float16* w1t  = (_Float16*)(ws + alloc(256 * 256 * 2));
  _Float16* w2t  = (_Float16*)(ws + alloc(128 * 256 * 2));
  float*    deg  = (float*)   (ws + alloc((size_t)N * 4));
  float*    dinv = (float*)   (ws + alloc((size_t)N * 4));
  _Float16* hbuf = (_Float16*)(ws + alloc((size_t)N * 256 * 2));   // h / feat1_f16 / feat2_f32
  float*    gbuf = (float*)   (ws + alloc((size_t)N * 256 * 4));   // gemm out (g1 / g2)
  float*    abuf = (float*)   (ws + alloc((size_t)N * 256 * 4));   // aggregation acc

  // prep
  pack_lstm<<<(1024 * LSTM_K + 255) / 256, 256, 0, stream>>>(w_ih, w_hh, b_ih, b_hh, wcat, bias);
  pack_transpose<<<(256 * 256 + 255) / 256, 256, 0, stream>>>(w1, w1t, 256, 256);
  pack_transpose<<<(256 * 128 + 255) / 256, 256, 0, stream>>>(w2, w2t, 256, 128);

  // LSTM (persistent-node, double-buffered LDS A-panel)
  {
    int blocks = (N + BLK_ROWS - 1) / BLK_ROWS;
    size_t lds = (size_t)2 * BLK_ROWS * A_STRIDE * sizeof(_Float16);
    lstm_kernel<<<blocks, 256, lds, stream>>>(x, wcat, bias, hbuf, N);
  }

  // degrees
  deg_init<<<(N + 255) / 256, 256, 0, stream>>>(deg, N);
  deg_count<<<(E + 255) / 256, 256, 0, stream>>>(dst, deg, E);
  dinv_k<<<(N + 255) / 256, 256, 0, stream>>>(deg, dinv, N);

  const int ntiles = (N + 15) / 16;
  const int gx = (ntiles + 3) / 4;
  long long tot1 = (long long)N * 256, tot2 = (long long)N * 128;

  // GCN conv 1: 256 -> 256
  gemm_f16<<<dim3(gx, 16), 128, 0, stream>>>(hbuf, w1t, gbuf, N, 256, 256);
  agg_init<<<(int)((tot1 + 255) / 256), 256, 0, stream>>>(gbuf, dinv, abuf, tot1, 256);
  agg_edge<<<(int)(((long long)E * 32 + 255) / 256), 256, 0, stream>>>(src, dst, dinv, gbuf, abuf, E, 256);
  finalize_f16<<<(int)((tot1 + 255) / 256), 256, 0, stream>>>(abuf, dinv, b1, hbuf, tot1, 256);

  // GCN conv 2: 256 -> 128
  gemm_f16<<<dim3(gx, 8), 128, 0, stream>>>(hbuf, w2t, gbuf, N, 256, 128);
  agg_init<<<(int)((tot2 + 255) / 256), 256, 0, stream>>>(gbuf, dinv, abuf, tot2, 128);
  agg_edge<<<(int)(((long long)E * 16 + 255) / 256), 256, 0, stream>>>(src, dst, dinv, gbuf, abuf, E, 128);
  finalize_f32<<<(int)((tot2 + 255) / 256), 256, 0, stream>>>(abuf, dinv, b2, (float*)hbuf, tot2, 128);

  // classifier + softmax
  classifier_k<<<(N + 255) / 256, 256, 0, stream>>>((const float*)hbuf, wc, bc, (float*)d_out, N);
}
